// ConvBlockE3_23424751632785
// MI455X (gfx1250) — matrix operand order repaired and checked by reading, not compiled, not run on hardware
//
#include <hip/hip_runtime.h>

typedef __attribute__((ext_vector_type(16))) _Float16 v16h;
typedef __attribute__((ext_vector_type(8)))  float    v8f;

#define INV_SQRT3  0.5773502691896258f
#define INV_SQRT32 0.17677669529663687f

#define NN 50000
#define NE 500000

#define DEVINL __device__ __forceinline__

// ---------------- LDS weight offsets (in _Float16 units), weights stored N-major (transposed) ----
enum {
  W_NM0 = 0,        // 64x64
  W_NM1 = 4096,     // 64x64
  W_NM2 = 8192,     // 64x96
  W_EM0 = 14336,
  W_EM1 = 18432,
  W_EM2 = 22528,
  W_NWA = 28672,    // 64x32
  W_NWB = 30720,    // 32x32
  W_NWC = 31744,    // 64x16
  W_NWD = 32768,    // 32x16
  W_EWA = 33280,    // 32x32
  W_EWB = 34304,    // 16x32
  W_EWC = 34816,    // 32x16
  W_EWD = 35328,    // 16x16
  W_NLS = 35584,    // 64x32
  W_NLV = 37632,    // 32x16
  W_ELS = 38144,    // 64x32
  W_ELV = 40192,    // 32x16
  W_ELOS = 40704,   // 32x32
  W_ELOV = 41728,   // 16x16
  W_TOTAL = 41984
};
#define W_BYTES (W_TOTAL * 2)           // 83968 bytes
#define PER_WAVE 20864                  // per-wave staging bytes
#define EDGE_WAVES 8
#define EDGE_SMEM (W_BYTES + EDGE_WAVES * PER_WAVE)   // 250880 bytes (< 320KB WGP LDS)
#define NODE_SMEM (2560 + 8 * 1024)

// ---------------- helpers -----------------------------------------------------------------------
DEVINL v8f vzero() {
  v8f z;
#pragma unroll
  for (int i = 0; i < 8; ++i) z[i] = 0.0f;
  return z;
}

// A fragment: 16 x 32 f16 tile (row-major in LDS, leading dim = lda), k window [koff, koff+32),
// zero-padded past kdepth. wave32 layout: lanes 0-15 rows M=lane, K 0-7 & 16-23;
// lanes 16-31 rows M=lane-16, K 8-15 & 24-31.
DEVINL v16h load_A(const _Float16* buf, int lda, int koff, int kdepth) {
  int l = threadIdx.x & 31;
  int m = l & 15;
  int kb = (l >> 4) << 3;     // 0 or 8
  v16h a;
#pragma unroll
  for (int e = 0; e < 8; ++e) {
    int k0 = kb + e;
    int k1 = 16 + kb + e;
    a[e]     = (k0 < kdepth) ? buf[m * lda + koff + k0] : (_Float16)0.0f;
    a[e + 8] = (k1 < kdepth) ? buf[m * lda + koff + k1] : (_Float16)0.0f;
  }
  return a;
}

// B fragment from N-major (transposed) weight wt[n*K + k]: lanes 0-15 K 0-7 & 16-23, lanes 16-31 K 8-15 & 24-31.
DEVINL v16h load_Bt(const _Float16* wt, int K, int col0, int koff, int kdepth) {
  int l = threadIdx.x & 31;
  int n = (l & 15) + col0;
  int kb = (l >> 4) << 3;     // 0 or 8
  v16h b;
#pragma unroll
  for (int e = 0; e < 16; ++e) {
    int k = (e < 8) ? (kb + e) : (16 + kb + e - 8);
    b[e] = (k < kdepth) ? wt[n * K + koff + k] : (_Float16)0.0f;
  }
  return b;
}

DEVINL v8f wmma_f16(v16h a, v16h b, v8f c) {
  return __builtin_amdgcn_wmma_f32_16x16x32_f16(false, a, false, b, (short)0, c, false, false);
}

// C = A(16xK) @ W(KxNOUT), W staged N-major with leading dim K. acc[NOUT/16] accumulators.
template<int K, int NOUT>
DEVINL void gemm(const _Float16* A, int lda, const _Float16* Wt, v8f* acc) {
#pragma unroll
  for (int kt = 0; kt < K; kt += 32) {
    v16h a = load_A(A, lda, kt, K - kt);
#pragma unroll
    for (int nt = 0; nt < NOUT / 16; ++nt) {
      v16h b = load_Bt(Wt, K, nt * 16, kt, K - kt);
      acc[nt] = wmma_f16(a, b, acc[nt]);
    }
  }
}

// accumulator stores (C/D layout: lanes 0-15 rows 0-7, lanes 16-31 rows 8-15, n = lane&15)
DEVINL void store_silu_f16(v8f acc, float scale, _Float16* out, int stride, int col0) {
  int l = threadIdx.x & 31;
  int n = (l & 15) + col0;
  int mb = (l >> 4) << 3;
#pragma unroll
  for (int r = 0; r < 8; ++r) {
    float h = acc[r] * scale;
    out[(mb + r) * stride + n] = (_Float16)(h / (1.0f + __expf(-h)));
  }
}

DEVINL void store_f32(v8f acc, float scale, float* out, int stride, int col0) {
  int l = threadIdx.x & 31;
  int n = (l & 15) + col0;
  int mb = (l >> 4) << 3;
#pragma unroll
  for (int r = 0; r < 8; ++r) out[(mb + r) * stride + n] = acc[r] * scale;
}

DEVINL void store_f16(v8f acc, float scale, _Float16* out, int stride, int col0) {
  int l = threadIdx.x & 31;
  int n = (l & 15) + col0;
  int mb = (l >> 4) << 3;
#pragma unroll
  for (int r = 0; r < 8; ++r) out[(mb + r) * stride + n] = (_Float16)(acc[r] * scale);
}

DEVINL void store_mul_f16(v8f acc, float scale, const float* wbuf, int wstride, int wcol0,
                          _Float16* out, int stride, int col0) {
  int l = threadIdx.x & 31;
  int n = l & 15;
  int mb = (l >> 4) << 3;
#pragma unroll
  for (int r = 0; r < 8; ++r) {
    int m = mb + r;
    out[m * stride + col0 + n] = (_Float16)(acc[r] * scale * wbuf[m * wstride + wcol0 + n]);
  }
}

DEVINL void atomic_rows(v8f acc, float scale, float* base, const int* rows, int rstride, int col0) {
  int l = threadIdx.x & 31;
  int n = (l & 15) + col0;
  int mb = (l >> 4) << 3;
#pragma unroll
  for (int r = 0; r < 8; ++r)
    atomicAdd(base + (size_t)rows[mb + r] * rstride + n, acc[r] * scale);
}

DEVINL void atomic_vec(v8f acc, float scale, float* base, const int* rows, int comp) {
  int l = threadIdx.x & 31;
  int n = l & 15;
  int mb = (l >> 4) << 3;
#pragma unroll
  for (int r = 0; r < 8; ++r)
    atomicAdd(base + (size_t)rows[mb + r] * 48 + n * 3 + comp, acc[r] * scale);
}

DEVINL void store_out_s(v8f acc, float scale, float* out, int row0, int col0) {
  int l = threadIdx.x & 31;
  int n = (l & 15) + col0;
  int mb = (l >> 4) << 3;
#pragma unroll
  for (int r = 0; r < 8; ++r) out[(size_t)(row0 + mb + r) * 80 + n] = acc[r] * scale;
}

DEVINL void store_out_v(v8f acc, float scale, float* out, int row0, int comp) {
  int l = threadIdx.x & 31;
  int n = l & 15;
  int mb = (l >> 4) << 3;
#pragma unroll
  for (int r = 0; r < 8; ++r)
    out[(size_t)(row0 + mb + r) * 80 + 32 + n * 3 + comp] = acc[r] * scale;
}

// stage f32 weight [K][N] row-major -> LDS f16 N-major wt[n*K + k]
DEVINL void stage_wt(_Float16* dst, const float* __restrict__ src, int K, int N) {
  for (int i = threadIdx.x; i < K * N; i += blockDim.x) {
    int k = i / N, n = i - k * N;
    dst[n * K + k] = (_Float16)src[i];
  }
}

// 3-layer radial MLP: 64 -> 64 (silu) -> 64 (silu) -> 96, all scales 1/8. Result f32 in outW[16][96].
DEVINL void radial_mlp(const _Float16* X, const _Float16* sW, int w0, int w1, int w2,
                       _Float16* tP, _Float16* tQ, float* outW) {
  v8f acc4[4];
#pragma unroll
  for (int t = 0; t < 4; ++t) acc4[t] = vzero();
  gemm<64, 64>(X, 64, sW + w0, acc4);
#pragma unroll
  for (int t = 0; t < 4; ++t) store_silu_f16(acc4[t], 0.125f, tP, 64, t * 16);
  __builtin_amdgcn_wave_barrier();
#pragma unroll
  for (int t = 0; t < 4; ++t) acc4[t] = vzero();
  gemm<64, 64>(tP, 64, sW + w1, acc4);
#pragma unroll
  for (int t = 0; t < 4; ++t) store_silu_f16(acc4[t], 0.125f, tQ, 64, t * 16);
  __builtin_amdgcn_wave_barrier();
  v8f acc6[6];
#pragma unroll
  for (int t = 0; t < 6; ++t) acc6[t] = vzero();
  gemm<64, 96>(tQ, 64, sW + w2, acc6);
#pragma unroll
  for (int t = 0; t < 6; ++t) store_f32(acc6[t], 0.125f, outW, 96, t * 16);
  __builtin_amdgcn_wave_barrier();
}

// ---------------- kernels -----------------------------------------------------------------------
struct EdgeParams {
  const float* __restrict__ node_feats;
  const float* __restrict__ edge_feats;
  const float* __restrict__ edge_sh;
  const float* __restrict__ edge_scalars;
  const int* __restrict__ src;
  const int* __restrict__ dst;
  const float *nWa, *nWb, *nWc, *nWd, *eWa, *eWb, *eWc, *eWd;
  const float *nM0, *nM1, *nM2, *eM0, *eM1, *eM2;
  const float *nLs, *nLv, *eLs, *eLv, *eLOs, *eLOv;
  float* __restrict__ aggS;
  float* __restrict__ aggV;
  float* __restrict__ edge_out;
};

__global__ void zero_kernel(float* p, int n) {
  int i = blockIdx.x * blockDim.x + threadIdx.x;
  if (i < n) p[i] = 0.0f;
}

__global__ __launch_bounds__(256, 1) void conv_edge_kernel(EdgeParams p, int nTiles) {
  extern __shared__ char smem[];
  _Float16* sW = (_Float16*)smem;

  stage_wt(sW + W_NM0, p.nM0, 64, 64);  stage_wt(sW + W_NM1, p.nM1, 64, 64);
  stage_wt(sW + W_NM2, p.nM2, 64, 96);  stage_wt(sW + W_EM0, p.eM0, 64, 64);
  stage_wt(sW + W_EM1, p.eM1, 64, 64);  stage_wt(sW + W_EM2, p.eM2, 64, 96);
  stage_wt(sW + W_NWA, p.nWa, 64, 32);  stage_wt(sW + W_NWB, p.nWb, 32, 32);
  stage_wt(sW + W_NWC, p.nWc, 64, 16);  stage_wt(sW + W_NWD, p.nWd, 32, 16);
  stage_wt(sW + W_EWA, p.eWa, 32, 32);  stage_wt(sW + W_EWB, p.eWb, 16, 32);
  stage_wt(sW + W_EWC, p.eWc, 32, 16);  stage_wt(sW + W_EWD, p.eWd, 16, 16);
  stage_wt(sW + W_NLS, p.nLs, 64, 32);  stage_wt(sW + W_NLV, p.nLv, 32, 16);
  stage_wt(sW + W_ELS, p.eLs, 64, 32);  stage_wt(sW + W_ELV, p.eLv, 32, 16);
  stage_wt(sW + W_ELOS, p.eLOs, 32, 32); stage_wt(sW + W_ELOV, p.eLOv, 16, 16);
  __syncthreads();

  int wave = threadIdx.x >> 5;
  int lane = threadIdx.x & 31;
  char* wb = smem + W_BYTES + wave * PER_WAVE;
  _Float16* bufX = (_Float16*)(wb);             // 16x64 f16: edge_scalars (persistent per tile)
  _Float16* bufP = (_Float16*)(wb + 2048);      // 16x64 f16 scratch
  _Float16* bufQ = (_Float16*)(wb + 4096);      // 16x64 f16 scratch
  _Float16* bufC = (_Float16*)(wb + 6144);      // 16x64 f16 scratch
  float*    bufW = (float*)(wb + 8192);         // 16x96 f32: radial weights
  float*    bufV = (float*)(wb + 14336);        // 16x3x32 f32: vector feats
  float*    bufSH = (float*)(wb + 20480);       // 16x4 f32
  int*      idxS = (int*)(wb + 20736);
  int*      idxD = (int*)(wb + 20800);

  int gw = blockIdx.x * EDGE_WAVES + wave;
  int nw = gridDim.x * EDGE_WAVES;

  for (int tile = gw; tile < nTiles; tile += nw) {
    int e0 = tile * 16;

    // prefetch next tile's streaming rows (emits global_prefetch_b8 on gfx1250)
    if (tile + nw < nTiles) {
      int en = (tile + nw) * 16 + (lane & 15);
      __builtin_prefetch(&p.edge_scalars[(size_t)en * 64], 0, 3);
      __builtin_prefetch(&p.edge_scalars[(size_t)en * 64 + 32], 0, 3);
      __builtin_prefetch(&p.edge_feats[(size_t)en * 80], 0, 3);
      __builtin_prefetch(&p.edge_feats[(size_t)en * 80 + 48], 0, 3);
    }

    if (lane < 16) { idxS[lane] = p.src[e0 + lane]; idxD[lane] = p.dst[e0 + lane]; }
    for (int t = lane; t < 64; t += 32) bufSH[t] = p.edge_sh[(size_t)e0 * 4 + t];
    __builtin_amdgcn_wave_barrier();
    for (int t = lane; t < 1024; t += 32) {
      int m = t >> 6, c = t & 63;
      bufX[t] = (_Float16)p.edge_scalars[(size_t)(e0 + m) * 64 + c];
    }
    __builtin_amdgcn_wave_barrier();

    // ======== radial MLP (node weights) -> bufW = wn ========
    radial_mlp(bufX, sW, W_NM0, W_NM1, W_NM2, bufP, bufQ, bufW);

    // ======== node message path ========
    for (int t = lane; t < 1024; t += 32) {   // gather cs -> bufC ; cs*sh0 -> bufP
      int m = t >> 6, c = t & 63;
      int nd = (c < 32) ? idxS[m] : idxD[m];
      float v = p.node_feats[(size_t)nd * 80 + (c & 31)];
      bufC[t] = (_Float16)v;
      bufP[t] = (_Float16)(v * bufSH[m * 4]);
    }
    for (int t = lane; t < 1536; t += 32) {   // gather cv -> bufV[m][i][u]
      int m = t / 96; int rr = t - m * 96; int i = rr >> 5; int u = rr & 31;
      int nd = (u < 16) ? idxS[m] : idxD[m];
      bufV[m * 96 + i * 32 + u] = p.node_feats[(size_t)nd * 80 + 32 + (u & 15) * 3 + i];
    }
    __builtin_amdgcn_wave_barrier();
    for (int t = lane; t < 512; t += 32) {    // dot(cv, shv) -> bufQ
      int m = t >> 5, u = t & 31;
      float d = bufV[m * 96 + u] * bufSH[m * 4 + 1] + bufV[m * 96 + 32 + u] * bufSH[m * 4 + 2] +
                bufV[m * 96 + 64 + u] * bufSH[m * 4 + 3];
      bufQ[m * 64 + u] = (_Float16)(d * INV_SQRT3);
    }
    __builtin_amdgcn_wave_barrier();
    v8f sacc[4];
#pragma unroll
    for (int t = 0; t < 4; ++t) sacc[t] = vzero();
    gemm<64, 32>(bufP, 64, sW + W_NWA, sacc);       // sa
    gemm<32, 32>(bufQ, 64, sW + W_NWB, sacc + 2);   // sb
    v8f vp = vzero();
    gemm<64, 16>(bufC, 64, sW + W_NWC, &vp);        // vc pre-factor
    __builtin_amdgcn_wave_barrier();
    store_mul_f16(sacc[0], 0.125f, bufW, 96, 0, bufP, 64, 0);     // ss = nms * wn[:, :64]
    store_mul_f16(sacc[1], 0.125f, bufW, 96, 16, bufP, 64, 16);
    store_mul_f16(sacc[2], INV_SQRT32, bufW, 96, 32, bufP, 64, 32);
    store_mul_f16(sacc[3], INV_SQRT32, bufW, 96, 48, bufP, 64, 48);
    __builtin_amdgcn_wave_barrier();
    v8f macc[2] = {vzero(), vzero()};
    gemm<64, 32>(bufP, 64, sW + W_NLS, macc);       // msg_s
    atomic_rows(macc[0], 0.125f, p.aggS, idxD, 32, 0);
    atomic_rows(macc[1], 0.125f, p.aggS, idxD, 32, 16);
#pragma unroll
    for (int i = 0; i < 3; ++i) {
      for (int t = lane; t < 512; t += 32) {        // A = cv_i * sh0
        int m = t >> 5, u = t & 31;
        bufQ[m * 64 + u] = (_Float16)(bufV[m * 96 + i * 32 + u] * bufSH[m * 4]);
      }
      __builtin_amdgcn_wave_barrier();
      v8f vd = vzero();
      gemm<32, 16>(bufQ, 64, sW + W_NWD, &vd);
      {
        int n = lane & 15, mb = (lane >> 4) << 3;   // vv_i = nmv_i * wn[:, 64:96] -> bufC 16x32
#pragma unroll
        for (int r = 0; r < 8; ++r) {
          int m = mb + r;
          bufC[m * 64 + n]      = (_Float16)(vp[r] * 0.125f * bufSH[m * 4 + 1 + i] * bufW[m * 96 + 64 + n]);
          bufC[m * 64 + 16 + n] = (_Float16)(vd[r] * INV_SQRT32 * bufW[m * 96 + 80 + n]);
        }
      }
      __builtin_amdgcn_wave_barrier();
      v8f vo = vzero();
      gemm<32, 16>(bufC, 64, sW + W_NLV, &vo);      // msg_v comp i
      atomic_vec(vo, INV_SQRT32, p.aggV, idxD, i);
      __builtin_amdgcn_wave_barrier();
    }

    // ======== radial MLP (edge weights) -> bufW = we ========
    radial_mlp(bufX, sW, W_EM0, W_EM1, W_EM2, bufP, bufQ, bufW);

    // ======== edge feature path ========
    for (int t = lane; t < 512; t += 32) {          // es -> bufC ; es*sh0 -> bufP
      int m = t >> 5, c = t & 31;
      float v = p.edge_feats[(size_t)(e0 + m) * 80 + c];
      bufC[m * 64 + c] = (_Float16)v;
      bufP[m * 64 + c] = (_Float16)(v * bufSH[m * 4]);
    }
    for (int t = lane; t < 768; t += 32) {          // ev -> bufV[m][i][u], u<16
      int m = t / 48; int rr = t - m * 48; int i = rr >> 4; int u = rr & 15;
      bufV[m * 96 + i * 32 + u] = p.edge_feats[(size_t)(e0 + m) * 80 + 32 + u * 3 + i];
    }
    __builtin_amdgcn_wave_barrier();
    for (int t = lane; t < 256; t += 32) {          // dot(ev, shv) -> bufQ (16x16)
      int m = t >> 4, u = t & 15;
      float d = bufV[m * 96 + u] * bufSH[m * 4 + 1] + bufV[m * 96 + 32 + u] * bufSH[m * 4 + 2] +
                bufV[m * 96 + 64 + u] * bufSH[m * 4 + 3];
      bufQ[m * 64 + u] = (_Float16)(d * INV_SQRT3);
    }
    __builtin_amdgcn_wave_barrier();
#pragma unroll
    for (int t = 0; t < 4; ++t) sacc[t] = vzero();
    gemm<32, 32>(bufP, 64, sW + W_EWA, sacc);
    gemm<16, 32>(bufQ, 64, sW + W_EWB, sacc + 2);
    vp = vzero();
    gemm<32, 16>(bufC, 64, sW + W_EWC, &vp);
    __builtin_amdgcn_wave_barrier();
    store_mul_f16(sacc[0], INV_SQRT32, bufW, 96, 0, bufP, 64, 0);
    store_mul_f16(sacc[1], INV_SQRT32, bufW, 96, 16, bufP, 64, 16);
    store_mul_f16(sacc[2], 0.25f, bufW, 96, 32, bufP, 64, 32);
    store_mul_f16(sacc[3], 0.25f, bufW, 96, 48, bufP, 64, 48);
    __builtin_amdgcn_wave_barrier();
    macc[0] = vzero(); macc[1] = vzero();
    gemm<64, 32>(bufP, 64, sW + W_ELS, macc);       // edg_s
    store_f16(macc[0], 0.125f, bufQ, 64, 0);
    store_f16(macc[1], 0.125f, bufQ, 64, 16);
    __builtin_amdgcn_wave_barrier();
    v8f oacc[2] = {vzero(), vzero()};
    gemm<32, 32>(bufQ, 64, sW + W_ELOS, oacc);      // edge_out scalars
    store_out_s(oacc[0], INV_SQRT32, p.edge_out, e0, 0);
    store_out_s(oacc[1], INV_SQRT32, p.edge_out, e0, 16);
#pragma unroll
    for (int i = 0; i < 3; ++i) {
      for (int t = lane; t < 256; t += 32) {        // A = ev_i * sh0
        int m = t >> 4, u = t & 15;
        bufC[m * 64 + u] = (_Float16)(bufV[m * 96 + i * 32 + u] * bufSH[m * 4]);
      }
      __builtin_amdgcn_wave_barrier();
      v8f vd = vzero();
      gemm<16, 16>(bufC, 64, sW + W_EWD, &vd);
      {
        int n = lane & 15, mb = (lane >> 4) << 3;   // vv_i -> bufP
#pragma unroll
        for (int r = 0; r < 8; ++r) {
          int m = mb + r;
          bufP[m * 64 + n]      = (_Float16)(vp[r] * INV_SQRT32 * bufSH[m * 4 + 1 + i] * bufW[m * 96 + 64 + n]);
          bufP[m * 64 + 16 + n] = (_Float16)(vd[r] * 0.25f * bufW[m * 96 + 80 + n]);
        }
      }
      __builtin_amdgcn_wave_barrier();
      v8f vo = vzero();
      gemm<32, 16>(bufP, 64, sW + W_ELV, &vo);      // edg_v comp i
      store_f16(vo, INV_SQRT32, bufQ, 64, 0);
      __builtin_amdgcn_wave_barrier();
      v8f ov = vzero();
      gemm<16, 16>(bufQ, 64, sW + W_ELOV, &ov);     // edge_out vector comp i
      store_out_v(ov, 0.25f, p.edge_out, e0, i);
      __builtin_amdgcn_wave_barrier();
    }
  }
}

__global__ __launch_bounds__(256, 1) void conv_node_kernel(const float* __restrict__ aggS,
                                                           const float* __restrict__ aggV,
                                                           const float* __restrict__ nLOs,
                                                           const float* __restrict__ nLOv,
                                                           float* __restrict__ node_out, int nTiles) {
  extern __shared__ char smem[];
  _Float16* sOs = (_Float16*)smem;        // 32x32 (N-major)
  _Float16* sOv = sOs + 1024;             // 16x16 (N-major)
  stage_wt(sOs, nLOs, 32, 32);
  stage_wt(sOv, nLOv, 16, 16);
  __syncthreads();
  int wave = threadIdx.x >> 5, lane = threadIdx.x & 31;
  _Float16* bufA = (_Float16*)(smem + 2560) + wave * 512;   // 16x32 f16 per wave
  int gw = blockIdx.x * 8 + wave, nw = gridDim.x * 8;
  for (int tile = gw; tile < nTiles; tile += nw) {
    int n0 = tile * 16;
    for (int t = lane; t < 512; t += 32) {
      int m = t >> 5, c = t & 31;
      bufA[t] = (_Float16)aggS[(size_t)(n0 + m) * 32 + c];
    }
    __builtin_amdgcn_wave_barrier();
    v8f acc[2] = {vzero(), vzero()};
    gemm<32, 32>(bufA, 32, sOs, acc);
    store_out_s(acc[0], INV_SQRT32, node_out, n0, 0);
    store_out_s(acc[1], INV_SQRT32, node_out, n0, 16);
#pragma unroll
    for (int i = 0; i < 3; ++i) {
      __builtin_amdgcn_wave_barrier();
      for (int t = lane; t < 256; t += 32) {
        int m = t >> 4, u = t & 15;
        bufA[m * 32 + u] = (_Float16)aggV[(size_t)(n0 + m) * 48 + u * 3 + i];
      }
      __builtin_amdgcn_wave_barrier();
      v8f vo = vzero();
      gemm<16, 16>(bufA, 32, sOv, &vo);
      store_out_v(vo, 0.25f, node_out, n0, i);
    }
    __builtin_amdgcn_wave_barrier();
  }
}

// ---------------- host side ---------------------------------------------------------------------
extern "C" void kernel_launch(void* const* d_in, const int* in_sizes, int n_in,
                              void* d_out, int out_size, void* d_ws, size_t ws_size,
                              hipStream_t stream) {
  (void)in_sizes; (void)n_in; (void)out_size; (void)ws_size;

  EdgeParams p;
  p.node_feats   = (const float*)d_in[0];
  p.edge_feats   = (const float*)d_in[1];
  p.edge_sh      = (const float*)d_in[2];
  p.edge_scalars = (const float*)d_in[3];
  p.src          = (const int*)d_in[4];
  p.dst          = (const int*)d_in[5];
  // params dict order: n_Wa..n_Wd, e_Wa..e_Wd, n_mlp[3], e_mlp[3], n_Ls,n_Lv,e_Ls,e_Lv,n_LOs,n_LOv,e_LOs,e_LOv
  p.nWa = (const float*)d_in[6];  p.nWb = (const float*)d_in[7];
  p.nWc = (const float*)d_in[8];  p.nWd = (const float*)d_in[9];
  p.eWa = (const float*)d_in[10]; p.eWb = (const float*)d_in[11];
  p.eWc = (const float*)d_in[12]; p.eWd = (const float*)d_in[13];
  p.nM0 = (const float*)d_in[14]; p.nM1 = (const float*)d_in[15]; p.nM2 = (const float*)d_in[16];
  p.eM0 = (const float*)d_in[17]; p.eM1 = (const float*)d_in[18]; p.eM2 = (const float*)d_in[19];
  p.nLs = (const float*)d_in[20]; p.nLv = (const float*)d_in[21];
  p.eLs = (const float*)d_in[22]; p.eLv = (const float*)d_in[23];
  const float* nLOs = (const float*)d_in[24];
  const float* nLOv = (const float*)d_in[25];
  p.eLOs = (const float*)d_in[26]; p.eLOv = (const float*)d_in[27];

  float* aggS = (float*)d_ws;                   // NN x 32
  float* aggV = aggS + (size_t)NN * 32;         // NN x 16 x 3
  p.aggS = aggS; p.aggV = aggV;

  float* node_out = (float*)d_out;              // NN x 80
  float* edge_out = node_out + (size_t)NN * 80; // NE x 80
  p.edge_out = edge_out;

  hipFuncSetAttribute(reinterpret_cast<const void*>(conv_edge_kernel),
                      hipFuncAttributeMaxDynamicSharedMemorySize, EDGE_SMEM);

  int nAgg = NN * 80;  // 32 + 48 floats per node
  zero_kernel<<<(nAgg + 255) / 256, 256, 0, stream>>>(aggS, nAgg);
  conv_edge_kernel<<<512, 256, EDGE_SMEM, stream>>>(p, NE / 16);
  conv_node_kernel<<<64, 256, NODE_SMEM, stream>>>(aggS, aggV, nLOs, nLOv, node_out, NN / 16);
}
